// GPTBigCodeAttention_58050777973165
// MI455X (gfx1250) — matrix operand order repaired and checked by reading, not compiled, not run on hardware
//
#include <hip/hip_runtime.h>
#include <hip/hip_bf16.h>

// ---------------------------------------------------------------------------
// CDNA5 (gfx1250) WMMA bf16 pipeline for GPTBigCode MQA attention block.
//   0) one-time f32 -> bf16 conversion of hs, c_attn_w, c_proj_w
//   1) qkv_bf  = hs_bf @ attn_w_bf^T + attn_b        (GEMM, bf16 WMMA, bf16 out)
//   2) ctx_bf  = causal-softmax(q k^T / sqrt(hd)) v  (flash MQA, bf16 WMMA)
//   3) out     = ctx_bf @ proj_w_bf^T + proj_b       (GEMM, bf16 WMMA, f32 out)
// GEMM tile staging uses GLOBAL_LOAD_ASYNC_TO_LDS_B128 (ASYNCcnt) with
// double-buffered LDS; __launch_bounds__(256,2) keeps the VGPR budget at
// file/2 to avoid scratch spills.
// ---------------------------------------------------------------------------

#if defined(__has_builtin)
#  if __has_builtin(__builtin_amdgcn_global_load_async_to_lds_b128) && \
      __has_builtin(__builtin_amdgcn_s_wait_asynccnt)
#    define USE_ASYNC_LDS 1
#  endif
#endif
#ifndef USE_ASYNC_LDS
#  define USE_ASYNC_LDS 0
#endif

#define AS1 __attribute__((address_space(1)))
#define AS3 __attribute__((address_space(3)))

typedef __bf16 bf16_t;
typedef __attribute__((ext_vector_type(8)))  bf16_t v8bf;
typedef __attribute__((ext_vector_type(16))) bf16_t v16bf;
typedef __attribute__((ext_vector_type(8)))  float  v8f;
typedef int v4i_t __attribute__((vector_size(16)));

static __device__ inline bf16_t f2bf(float f) {
  union { float f; unsigned u; } x; x.f = f;
  unsigned r = x.u + 0x7FFFu + ((x.u >> 16) & 1u);
  unsigned short h = (unsigned short)(r >> 16);
  union { unsigned short s; bf16_t b; } y; y.s = h;
  return y.b;
}

static __device__ inline unsigned pk2bf(float lo, float hi) {
  union { float f; unsigned u; } a, b; a.f = lo; b.f = hi;
  unsigned ra = (a.u + 0x7FFFu + ((a.u >> 16) & 1u)) >> 16;
  unsigned rb = (b.u + 0x7FFFu + ((b.u >> 16) & 1u)) & 0xFFFF0000u;
  return ra | rb;
}

static __device__ inline v16bf join8(v8bf lo, v8bf hi) {
  return __builtin_shufflevector(lo, hi, 0,1,2,3,4,5,6,7,8,9,10,11,12,13,14,15);
}

static __device__ inline v8f wmma_bf16(v16bf a, v16bf b, v8f c) {
  return __builtin_amdgcn_wmma_f32_16x16x32_bf16(false, a, false, b, (short)0, c,
                                                 false, false);
}

// A fragment (16x32 bf16). tile = row-major [16][K], row stride `ld` elements.
static __device__ inline v16bf lds_a_16x32(const bf16_t* tile, int ld) {
  const int lane = threadIdx.x & 31;
  const int m  = lane & 15;
  const int kb = (lane >> 4) * 8;
  const bf16_t* r = tile + m * ld + kb;
  return join8(*(const v8bf*)(r), *(const v8bf*)(r + 16));
}

// B fragment (32x16 bf16) with B[k][n] stored at tile[n*ld + k].
static __device__ inline v16bf lds_b_32x16(const bf16_t* tile, int ld) {
  const int lane = threadIdx.x & 31;
  const int n  = lane & 15;
  const int kb = (lane >> 4) * 8;
  const bf16_t* r = tile + n * ld + kb;
  return join8(*(const v8bf*)(r), *(const v8bf*)(r + 16));
}

#if USE_ASYNC_LDS
// Per-lane async DMA: 16 bytes global -> 16 bytes LDS, tracked by ASYNCcnt.
static __device__ inline void async_copy_b128(const bf16_t* g, bf16_t* l) {
  __builtin_amdgcn_global_load_async_to_lds_b128((AS1 v4i_t*)g, (AS3 v4i_t*)l,
                                                 0, 0);
}
#endif

// ---------------------------------------------------------------------------
// 0) f32 -> bf16 bulk conversion
// ---------------------------------------------------------------------------
__global__ __launch_bounds__(256)
void cvt_f32_bf16(const float* __restrict__ src, bf16_t* __restrict__ dst, size_t n) {
  const size_t i = ((size_t)blockIdx.x * 256 + threadIdx.x) * 8;
  if (i + 8 > n) return;
  const float4 a = *(const float4*)(src + i);
  const float4 b = *(const float4*)(src + i + 4);
  uint4 o;
  o.x = pk2bf(a.x, a.y); o.y = pk2bf(a.z, a.w);
  o.z = pk2bf(b.x, b.y); o.w = pk2bf(b.z, b.w);
  *(uint4*)(dst + i) = o;
}

// ---------------------------------------------------------------------------
// GEMM: C[M,N] = A[M,K] @ W[N,K]^T + bias[N]  (bf16 in, f32 accum)
// Out: bf16 (Cb != nullptr) or f32 (Cf). Block 128x128, K-step 64,
// 8 waves (4M x 2N), 32x64 per wave.
// ---------------------------------------------------------------------------
__global__ __launch_bounds__(256, 2)
void gemm_bf16wmma(const bf16_t* __restrict__ A, const bf16_t* __restrict__ W,
                   const float* __restrict__ bias, float* __restrict__ Cf,
                   bf16_t* __restrict__ Cb, int M, int N, int K) {
#if USE_ASYNC_LDS
  __shared__ __attribute__((aligned(16))) bf16_t Alds[2][128 * 64];
  __shared__ __attribute__((aligned(16))) bf16_t Wlds[2][128 * 64];
#else
  __shared__ __attribute__((aligned(16))) bf16_t Alds[1][128 * 64];
  __shared__ __attribute__((aligned(16))) bf16_t Wlds[1][128 * 64];
#endif

  const int tid  = threadIdx.x;
  const int lane = tid & 31;
  const int wid  = tid >> 5;
  const int n0   = blockIdx.x * 128;
  const int m0   = blockIdx.y * 128;
  const int wm   = (wid & 3) * 32;
  const int wn   = (wid >> 2) * 64;

  // this thread's 4 staging slots (8 bf16 each) inside the 128x64 tile
  int srow[4], scol[4];
  #pragma unroll
  for (int v = 0; v < 4; ++v) {
    const int e = (tid + v * 256) * 8;   // 0..8184
    srow[v] = e >> 6;
    scol[v] = e & 63;
  }

  v8f acc[2][4];
  v8f zero = {};
  #pragma unroll
  for (int i = 0; i < 2; ++i)
    #pragma unroll
    for (int j = 0; j < 4; ++j) acc[i][j] = zero;

  const int NT = K >> 6;

#if USE_ASYNC_LDS
  // ---- async double-buffered staging: no VGPR staging at all ----
  #pragma unroll
  for (int v = 0; v < 4; ++v) {
    async_copy_b128(A + (size_t)(m0 + srow[v]) * K + scol[v],
                    &Alds[0][srow[v] * 64 + scol[v]]);
    async_copy_b128(W + (size_t)(n0 + srow[v]) * K + scol[v],
                    &Wlds[0][srow[v] * 64 + scol[v]]);
  }
  for (int kt = 0; kt < NT; ++kt) {
    __builtin_amdgcn_s_wait_asynccnt(0);  // this wave's DMA into cur buffer done
    __syncthreads();                      // everyone's DMA done
    const int cur = kt & 1;
    if (kt + 1 < NT) {
      const int k1  = (kt + 1) << 6;
      const int nxt = cur ^ 1;
      #pragma unroll
      for (int v = 0; v < 4; ++v) {
        async_copy_b128(A + (size_t)(m0 + srow[v]) * K + k1 + scol[v],
                        &Alds[nxt][srow[v] * 64 + scol[v]]);
        async_copy_b128(W + (size_t)(n0 + srow[v]) * K + k1 + scol[v],
                        &Wlds[nxt][srow[v] * 64 + scol[v]]);
      }
      if (kt + 2 < NT) {
        const int k2 = (kt + 2) << 6;
        __builtin_prefetch(A + (size_t)(m0 + srow[0]) * K + k2 + scol[0], 0, 1);
        __builtin_prefetch(W + (size_t)(n0 + srow[0]) * K + k2 + scol[0], 0, 1);
      }
    }
    const bf16_t* Ab = Alds[cur];
    const bf16_t* Wb = Wlds[cur];
    #pragma unroll
    for (int c = 0; c < 2; ++c) {
      const v16bf a0 = lds_a_16x32(&Ab[(wm +  0) * 64 + c * 32], 64);
      const v16bf a1 = lds_a_16x32(&Ab[(wm + 16) * 64 + c * 32], 64);
      #pragma unroll
      for (int t = 0; t < 4; ++t) {
        const v16bf b = lds_b_32x16(&Wb[(wn + t * 16) * 64 + c * 32], 64);
        acc[0][t] = wmma_bf16(a0, b, acc[0][t]);
        acc[1][t] = wmma_bf16(a1, b, acc[1][t]);
      }
    }
  }
#else
  // ---- fallback: register-pipelined staging ----
  uint4 areg[4], wreg[4];
  #pragma unroll
  for (int v = 0; v < 4; ++v) {
    areg[v] = *(const uint4*)(A + (size_t)(m0 + srow[v]) * K + scol[v]);
    wreg[v] = *(const uint4*)(W + (size_t)(n0 + srow[v]) * K + scol[v]);
  }
  for (int kt = 0; kt < NT; ++kt) {
    __syncthreads();
    #pragma unroll
    for (int v = 0; v < 4; ++v) {
      *(uint4*)&Alds[0][srow[v] * 64 + scol[v]] = areg[v];
      *(uint4*)&Wlds[0][srow[v] * 64 + scol[v]] = wreg[v];
    }
    __syncthreads();
    if (kt + 1 < NT) {
      const int k1 = (kt + 1) << 6;
      #pragma unroll
      for (int v = 0; v < 4; ++v) {
        areg[v] = *(const uint4*)(A + (size_t)(m0 + srow[v]) * K + k1 + scol[v]);
        wreg[v] = *(const uint4*)(W + (size_t)(n0 + srow[v]) * K + k1 + scol[v]);
      }
      if (kt + 2 < NT) {
        const int k2 = (kt + 2) << 6;
        __builtin_prefetch(A + (size_t)(m0 + srow[0]) * K + k2 + scol[0], 0, 1);
        __builtin_prefetch(W + (size_t)(n0 + srow[0]) * K + k2 + scol[0], 0, 1);
      }
    }
    #pragma unroll
    for (int c = 0; c < 2; ++c) {
      const v16bf a0 = lds_a_16x32(&Alds[0][(wm +  0) * 64 + c * 32], 64);
      const v16bf a1 = lds_a_16x32(&Alds[0][(wm + 16) * 64 + c * 32], 64);
      #pragma unroll
      for (int t = 0; t < 4; ++t) {
        const v16bf b = lds_b_32x16(&Wlds[0][(wn + t * 16) * 64 + c * 32], 64);
        acc[0][t] = wmma_bf16(a0, b, acc[0][t]);
        acc[1][t] = wmma_bf16(a1, b, acc[1][t]);
      }
    }
  }
#endif

  const int coll = lane & 15;
  const int hi   = lane >> 4;
  if (Cb) {
    #pragma unroll
    for (int mt = 0; mt < 2; ++mt)
      #pragma unroll
      for (int nt = 0; nt < 4; ++nt) {
        const int col = n0 + wn + nt * 16 + coll;
        const float bv = bias[col];
        #pragma unroll
        for (int r = 0; r < 8; ++r) {
          const int row = m0 + wm + mt * 16 + hi * 8 + r;
          Cb[(size_t)row * N + col] = f2bf(acc[mt][nt][r] + bv);
        }
      }
  } else {
    #pragma unroll
    for (int mt = 0; mt < 2; ++mt)
      #pragma unroll
      for (int nt = 0; nt < 4; ++nt) {
        const int col = n0 + wn + nt * 16 + coll;
        const float bv = bias[col];
        #pragma unroll
        for (int r = 0; r < 8; ++r) {
          const int row = m0 + wm + mt * 16 + hi * 8 + r;
          Cf[(size_t)row * N + col] = acc[mt][nt][r] + bv;
        }
      }
  }
  (void)M;
}

// ---------------------------------------------------------------------------
// Flash-style causal multi-query attention (all-bf16 operands).
// ---------------------------------------------------------------------------
__global__ __launch_bounds__(256, 2)
void flash_mqa(const bf16_t* __restrict__ qkv, bf16_t* __restrict__ ctx, int S) {
  constexpr int HD   = 128;
  constexpr int QKVW = 2304;
  constexpr int D    = 2048;
  constexpr float SCALE = 0.08838834764831845f;  // 1/sqrt(128)

  __shared__ __attribute__((aligned(16))) bf16_t Klds[32 * 128];     // [key][dim]
  __shared__ __attribute__((aligned(16))) bf16_t Vt[128 * 32];       // [dim][key]
  __shared__ __attribute__((aligned(16))) bf16_t Plds[8 * 16 * 32];  // per-wave P

  const int tid  = threadIdx.x;
  const int lane = tid & 31;
  const int wid  = tid >> 5;
  const int q0   = blockIdx.x * 128;
  const int h    = blockIdx.y;
  const int b    = blockIdx.z;
  const int wq0  = q0 + wid * 16;
  const int coll = lane & 15;
  const int hi   = lane >> 4;

  v16bf qf[4];
  {
    const bf16_t* qr = qkv + ((size_t)b * S + wq0 + coll) * QKVW + h * HD + hi * 8;
    #pragma unroll
    for (int c = 0; c < 4; ++c)
      qf[c] = join8(*(const v8bf*)(qr + c * 32), *(const v8bf*)(qr + c * 32 + 16));
  }

  int skey[2], sdim[2];
  #pragma unroll
  for (int v = 0; v < 2; ++v) {
    const int e = (tid + v * 256) * 8;
    skey[v] = e >> 7;
    sdim[v] = e & 127;
  }

  uint4 kreg[2], vreg[2];
  auto load_chunk = [&](int key0) {
    #pragma unroll
    for (int v = 0; v < 2; ++v) {
      const bf16_t* src =
          qkv + ((size_t)b * S + key0 + skey[v]) * QKVW + D + sdim[v];
      kreg[v] = *(const uint4*)src;
      vreg[v] = *(const uint4*)(src + HD);
    }
  };
  auto store_chunk = [&]() {
    #pragma unroll
    for (int v = 0; v < 2; ++v) {
      *(uint4*)&Klds[skey[v] * 128 + sdim[v]] = kreg[v];
      union { uint4 u; bf16_t h[8]; } vv; vv.u = vreg[v];
      #pragma unroll
      for (int i = 0; i < 8; ++i) Vt[(sdim[v] + i) * 32 + skey[v]] = vv.h[i];
    }
  };

  float m_r[8], l_r[8];
  #pragma unroll
  for (int r = 0; r < 8; ++r) { m_r[r] = -3.0e38f; l_r[r] = 0.0f; }
  v8f acco[8];
  v8f zero = {};
  #pragma unroll
  for (int n = 0; n < 8; ++n) acco[n] = zero;

  const int nchunk = q0 / 32 + 4;
  load_chunk(0);

  for (int j = 0; j < nchunk; ++j) {
    const int key0 = j * 32;
    __syncthreads();
    store_chunk();
    __syncthreads();
    if (j + 1 < nchunk) {
      load_chunk((j + 1) * 32);
      if (j + 2 < nchunk)
        __builtin_prefetch(
            qkv + ((size_t)b * S + (j + 2) * 32 + skey[0]) * QKVW + D + sdim[0],
            0, 1);
    }

    v8f s0 = zero, s1 = zero;
    #pragma unroll
    for (int c = 0; c < 4; ++c) {
      const v16bf bk0 = lds_b_32x16(Klds + 0 * 16 * 128 + c * 32, 128);
      const v16bf bk1 = lds_b_32x16(Klds + 1 * 16 * 128 + c * 32, 128);
      s0 = wmma_bf16(qf[c], bk0, s0);
      s1 = wmma_bf16(qf[c], bk1, s1);
    }

    float rmax[8];
    #pragma unroll
    for (int r = 0; r < 8; ++r) {
      const int row = wq0 + hi * 8 + r;
      float v0 = s0[r] * SCALE;
      float v1 = s1[r] * SCALE;
      if (key0 + coll > row)      v0 = -3.0e38f;
      if (key0 + 16 + coll > row) v1 = -3.0e38f;
      s0[r] = v0; s1[r] = v1;
      rmax[r] = fmaxf(v0, v1);
    }
    #pragma unroll
    for (int off = 1; off < 16; off <<= 1)
      #pragma unroll
      for (int r = 0; r < 8; ++r)
        rmax[r] = fmaxf(rmax[r], __shfl_xor(rmax[r], off, 32));

    float alpha[8], rsum[8];
    #pragma unroll
    for (int r = 0; r < 8; ++r) {
      const float mn = fmaxf(m_r[r], rmax[r]);
      alpha[r] = __expf(m_r[r] - mn);
      m_r[r] = mn;
      const float p0 = __expf(s0[r] - mn);
      const float p1 = __expf(s1[r] - mn);
      s0[r] = p0; s1[r] = p1;
      rsum[r] = p0 + p1;
    }
    #pragma unroll
    for (int off = 1; off < 16; off <<= 1)
      #pragma unroll
      for (int r = 0; r < 8; ++r)
        rsum[r] += __shfl_xor(rsum[r], off, 32);
    #pragma unroll
    for (int r = 0; r < 8; ++r) l_r[r] = l_r[r] * alpha[r] + rsum[r];
    #pragma unroll
    for (int n = 0; n < 8; ++n)
      #pragma unroll
      for (int r = 0; r < 8; ++r)
        acco[n][r] = acco[n][r] * alpha[r];

    bf16_t* pw = Plds + wid * 16 * 32;
    #pragma unroll
    for (int r = 0; r < 8; ++r) {
      pw[(hi * 8 + r) * 32 + coll]      = f2bf(s0[r]);
      pw[(hi * 8 + r) * 32 + 16 + coll] = f2bf(s1[r]);
    }
    __syncthreads();
    const v16bf pf = lds_a_16x32(pw, 32);

    #pragma unroll
    for (int n = 0; n < 8; ++n) {
      const v16bf bv = lds_b_32x16(Vt + n * 16 * 32, 32);
      acco[n] = wmma_bf16(pf, bv, acco[n]);
    }
  }

  float inv[8];
  #pragma unroll
  for (int r = 0; r < 8; ++r) inv[r] = 1.0f / l_r[r];
  #pragma unroll
  for (int n = 0; n < 8; ++n)
    #pragma unroll
    for (int r = 0; r < 8; ++r) {
      const int row = wq0 + hi * 8 + r;
      const int dim = n * 16 + coll;
      ctx[((size_t)b * S + row) * D + h * HD + dim] = f2bf(acco[n][r] * inv[r]);
    }
}

// ---------------------------------------------------------------------------
extern "C" void kernel_launch(void* const* d_in, const int* in_sizes, int n_in,
                              void* d_out, int out_size, void* d_ws, size_t ws_size,
                              hipStream_t stream) {
  (void)in_sizes; (void)n_in; (void)out_size; (void)ws_size;

  constexpr int B = 2, S = 2048, D = 2048, NQKV = 2304;
  constexpr size_t nHS  = (size_t)B * S * D;
  constexpr size_t nAW  = (size_t)NQKV * D;
  constexpr size_t nPW  = (size_t)D * D;
  constexpr size_t nQKV = (size_t)B * S * NQKV;

  const float* hs     = (const float*)d_in[0];
  const float* attn_w = (const float*)d_in[1];
  const float* attn_b = (const float*)d_in[2];
  const float* proj_w = (const float*)d_in[3];
  const float* proj_b = (const float*)d_in[4];
  float* out = (float*)d_out;

  bf16_t* hsb  = (bf16_t*)d_ws;
  bf16_t* awb  = hsb + nHS;
  bf16_t* pwb  = awb + nAW;
  bf16_t* qkvb = pwb + nPW;
  bf16_t* ctxb = qkvb + nQKV;

  cvt_f32_bf16<<<dim3((unsigned)(nHS / 2048)), 256, 0, stream>>>(hs, hsb, nHS);
  cvt_f32_bf16<<<dim3((unsigned)(nAW / 2048)), 256, 0, stream>>>(attn_w, awb, nAW);
  cvt_f32_bf16<<<dim3((unsigned)(nPW / 2048)), 256, 0, stream>>>(proj_w, pwb, nPW);

  {
    dim3 grid(NQKV / 128, (B * S) / 128);
    gemm_bf16wmma<<<grid, 256, 0, stream>>>(hsb, awb, attn_b, nullptr, qkvb,
                                            B * S, NQKV, D);
  }
  {
    dim3 grid(S / 128, 16, B);
    flash_mqa<<<grid, 256, 0, stream>>>(qkvb, ctxb, S);
  }
  {
    dim3 grid(D / 128, (B * S) / 128);
    gemm_bf16wmma<<<grid, 256, 0, stream>>>(ctxb, pwb, proj_b, out, nullptr,
                                            B * S, D, D);
  }
}
